// HHGR_82506321756638
// MI455X (gfx1250) — matrix operand order repaired and checked by reading, not compiled, not run on hardware
//
#include <hip/hip_runtime.h>
#include <hip/hip_bf16.h>

typedef __attribute__((ext_vector_type(16))) _Float16 v16h;
typedef __attribute__((ext_vector_type(8)))  _Float16 h8;
typedef __attribute__((ext_vector_type(4)))  _Float16 h4;
typedef __attribute__((ext_vector_type(8)))  float    v8f;

#define NW 8            // waves (=groups) per block
#define XSTR 72         // LDS row stride in halves for X (144B: 16B-aligned, bank-skewed)
#define W1TSTR 136      // LDS row stride for W1^T (272B: 16B-aligned, bank-skewed)

#define WMMA_F16(A, B, C) \
    __builtin_amdgcn_wmma_f32_16x16x32_f16(false, (A), false, (B), (short)0, (C), false, false)

__device__ __forceinline__ v16h pack16(h8 lo, h8 hi) {
    v16h t;
#pragma unroll
    for (int i = 0; i < 8; ++i) { t[i] = lo[i]; t[8 + i] = hi[i]; }
    return t;
}

__global__ __launch_bounds__(NW * 32) void hhgr_fused_kernel(
    const int* __restrict__ group_inputs,      // [B]
    const int* __restrict__ item_inputs,       // [B]
    const float* __restrict__ all_user_embeds, // [N_USERS,64]
    const float* __restrict__ all_group_embeds,// [N_GROUPS,64]
    const int* __restrict__ member_ids,        // [B*32]
    const float* __restrict__ user_emb,        // [N_USERS,64]
    const float* __restrict__ item_emb,        // [N_ITEMS,64]
    const float* __restrict__ group_emb,       // [N_GROUPS,64]
    const float* __restrict__ W1,              // [128,8] row-major
    const float* __restrict__ b1,              // [8]
    const float* __restrict__ W2,              // [8]
    const float* __restrict__ b2,              // [1]
    float* __restrict__ out,                   // [B]
    int B)
{
    __shared__ __align__(16) _Float16 s_w1t[16][W1TSTR];   // W1^T padded to 16 cols
    __shared__ __align__(16) _Float16 s_X[NW][32][XSTR];   // f16 member embeddings
    __shared__ __align__(16) _Float16 s_it[NW][64];        // f16 item embedding
    __shared__ float s_s[NW][32];                          // attention logits

    const int lane = threadIdx.x & 31;
    const int wv   = threadIdx.x >> 5;
    const int g    = blockIdx.x * NW + wv;

    // ---- stage W1^T (zero-padded N: 8->16) once per block ----
    for (int i = threadIdx.x; i < 16 * 128; i += blockDim.x) {
        int n = i >> 7, k = i & 127;
        float v = (n < 8) ? W1[k * 8 + n] : 0.0f;
        s_w1t[n][k] = (_Float16)v;
    }

    float it0 = 0.f, it1 = 0.f;
    if (g < B) {
        // ---- item embedding: lane owns dims {lane, lane+32} ----
        const int item = item_inputs[g];
        it0 = item_emb[(size_t)item * 64 + lane];
        it1 = item_emb[(size_t)item * 64 + 32 + lane];
        s_it[wv][lane]      = (_Float16)it0;
        s_it[wv][lane + 32] = (_Float16)it1;

        // ---- coalesced gather of 32 member rows (2 rows / iter, 16 lanes x float4) ----
        const int vec  = lane & 15;
        const int msel = lane >> 4;
#pragma unroll 4
        for (int it_i = 0; it_i < 16; ++it_i) {
            const int m   = it_i * 2 + msel;
            const int rid = member_ids[(size_t)g * 32 + m];
            const float4 u = ((const float4*)(user_emb         + (size_t)rid * 64))[vec];
            const float4 a = ((const float4*)(all_user_embeds  + (size_t)rid * 64))[vec];
            h4 hv;
            hv[0] = (_Float16)(u.x + a.x);
            hv[1] = (_Float16)(u.y + a.y);
            hv[2] = (_Float16)(u.z + a.z);
            hv[3] = (_Float16)(u.w + a.w);
            *(h4*)&s_X[wv][m][vec * 4] = hv;
        }
    }
    __syncthreads();

    if (g < B) {
        const int n  = lane & 15;
        const int hh = lane >> 4;

        // ---- B fragments: b[h] = W1T[n][kb*32 + 8*hh + h] (h<8), W1T[n][kb*32 + 16 + 8*hh + h-8] (h>=8) ----
        v16h bf[4];
#pragma unroll
        for (int kb = 0; kb < 4; ++kb) {
            const _Float16* p = &s_w1t[n][kb * 32 + hh * 8];
            bf[kb] = pack16(*(const h8*)p, *(const h8*)(p + 16));
        }

        // ---- A fragments, item half (row-independent, shared by both M-tiles) ----
        v16h ai[2];
#pragma unroll
        for (int kb = 0; kb < 2; ++kb) {
            const _Float16* p = &s_it[wv][kb * 32 + 8 * hh];
            ai[kb] = pack16(*(const h8*)p, *(const h8*)(p + 16));
        }

        // ---- A fragments, member half, for row tiles 0-15 and 16-31 ----
        v16h a0[2], a1[2];
#pragma unroll
        for (int kb = 0; kb < 2; ++kb) {
            const _Float16* p0 = &s_X[wv][n][kb * 32 + 8 * hh];
            const _Float16* p1 = &s_X[wv][16 + n][kb * 32 + 8 * hh];
            a0[kb] = pack16(*(const h8*)p0, *(const h8*)(p0 + 16));
            a1[kb] = pack16(*(const h8*)p1, *(const h8*)(p1 + 16));
        }

        // ---- h = X @ W1 : 8 WMMAs total ----
        v8f c0 = {}, c1 = {};
        c0 = WMMA_F16(a0[0], bf[0], c0);
        c0 = WMMA_F16(a0[1], bf[1], c0);
        c0 = WMMA_F16(ai[0], bf[2], c0);
        c0 = WMMA_F16(ai[1], bf[3], c0);
        c1 = WMMA_F16(a1[0], bf[0], c1);
        c1 = WMMA_F16(a1[1], bf[1], c1);
        c1 = WMMA_F16(ai[0], bf[2], c1);
        c1 = WMMA_F16(ai[1], bf[3], c1);

        // ---- bias + relu + @W2, reduce over N (16-lane halves) ----
        const float b1n = (n < 8) ? b1[n] : 0.0f;
        const float w2n = (n < 8) ? W2[n] : 0.0f;
        const float b2v = b2[0];
#pragma unroll
        for (int r = 0; r < 8; ++r) {
            float t0 = fmaxf(c0[r] + b1n, 0.0f) * w2n;
            float t1 = fmaxf(c1[r] + b1n, 0.0f) * w2n;
#pragma unroll
            for (int off = 1; off < 16; off <<= 1) {
                t0 += __shfl_xor(t0, off, 32);
                t1 += __shfl_xor(t1, off, 32);
            }
            if (n == r) {                       // unique writer per (tile,half)
                s_s[wv][r + 8 * hh]      = t0 + b2v;   // rows 0-15
                s_s[wv][16 + r + 8 * hh] = t1 + b2v;   // rows 16-31
            }
        }

        // ---- segment softmax over the 32 members (lane = member) ----
        const float sv = s_s[wv][lane];
        float mx = sv;
#pragma unroll
        for (int off = 16; off >= 1; off >>= 1) mx = fmaxf(mx, __shfl_xor(mx, off, 32));
        const float e = __expf(sv - mx);
        float den = e;
#pragma unroll
        for (int off = 16; off >= 1; off >>= 1) den += __shfl_xor(den, off, 32);
        const float wgt = e / den;

        // ---- weighted pooling: lane owns dims {lane, lane+32} ----
        float p0 = 0.f, p1 = 0.f;
#pragma unroll 8
        for (int m = 0; m < 32; ++m) {
            const float wm = __shfl(wgt, m, 32);
            p0 += (float)s_X[wv][m][lane] * wm;
            p1 += (float)s_X[wv][m][lane + 32] * wm;
        }

        // ---- add group embeddings, dot with item embedding ----
        const int gid = group_inputs[g];
        const float g0 = p0 + group_emb[(size_t)gid * 64 + lane]
                            + all_group_embeds[(size_t)gid * 64 + lane];
        const float g1 = p1 + group_emb[(size_t)gid * 64 + 32 + lane]
                            + all_group_embeds[(size_t)gid * 64 + 32 + lane];
        float dot = g0 * it0 + g1 * it1;
#pragma unroll
        for (int off = 16; off >= 1; off >>= 1) dot += __shfl_xor(dot, off, 32);
        if (lane == 0) out[g] = dot;
    }
}

extern "C" void kernel_launch(void* const* d_in, const int* in_sizes, int n_in,
                              void* d_out, int out_size, void* d_ws, size_t ws_size,
                              hipStream_t stream) {
    const int*   group_inputs     = (const int*)d_in[0];
    const int*   item_inputs      = (const int*)d_in[1];
    const float* all_user_embeds  = (const float*)d_in[2];
    const float* all_group_embeds = (const float*)d_in[3];
    const int*   member_ids       = (const int*)d_in[4];
    // d_in[5] = segment_ids: layout is repeat(arange(B), 32), derived implicitly
    const float* user_emb         = (const float*)d_in[6];
    const float* item_emb         = (const float*)d_in[7];
    const float* group_emb        = (const float*)d_in[8];
    const float* W1               = (const float*)d_in[9];
    const float* b1               = (const float*)d_in[10];
    const float* W2               = (const float*)d_in[11];
    const float* b2               = (const float*)d_in[12];
    float* out = (float*)d_out;

    const int B = in_sizes[0];
    const int blocks = (B + NW - 1) / NW;
    hipLaunchKernelGGL(hhgr_fused_kernel, dim3(blocks), dim3(NW * 32), 0, stream,
                       group_inputs, item_inputs, all_user_embeds, all_group_embeds,
                       member_ids, user_emb, item_emb, group_emb,
                       W1, b1, W2, b2, out, B);
}
